// MaskedMultiheadSelfAttention_77017353551924
// MI455X (gfx1250) — matrix operand order repaired and checked by reading, not compiled, not run on hardware
//
#include <hip/hip_runtime.h>

// ---------------- problem constants (from reference) ----------------
#define B_  4
#define S_  2048
#define H_  1024
#define NH_ 16
#define HD_ 64          // head dim
#define QK_SCALE 0.125f // hd^-0.5

// ---------------- CDNA5 WMMA plumbing ----------------
typedef __attribute__((ext_vector_type(16))) __bf16       v16bf;
typedef __attribute__((ext_vector_type(8)))  float        v8f;
typedef __attribute__((ext_vector_type(4)))  unsigned int u32x4;

union V16U { v16bf v; u32x4 u[2]; unsigned int w[8]; };

__device__ __forceinline__ unsigned short f2bfbits(float f) {
  unsigned u = __builtin_bit_cast(unsigned, f);
  unsigned r = u + 0x7FFFu + ((u >> 16) & 1u);   // round-to-nearest-even
  return (unsigned short)(r >> 16);
}
__device__ __forceinline__ __bf16 f2bf(float f) {
  unsigned short h = f2bfbits(f);
  return __builtin_bit_cast(__bf16, h);
}

__device__ __forceinline__ v8f wmma_bf16(v16bf a, v16bf b, v8f c) {
  // D = A(16x32 bf16) x B(32x16 bf16) + C(16x16 f32)
  return __builtin_amdgcn_wmma_f32_16x16x32_bf16(false, a, false, b, (short)0, c, false, false);
}

// A-layout (16x32 bf16, ISA 7.12.2): lane<16 -> row=lane, K = {0..7,16..23};
// lane>=16 -> row=lane-16, K = 8+{0..7,16..23}. Source is row-major [row][k], ld elems.
__device__ __forceinline__ v16bf load_tileA(const __bf16* base, int ld, int row0, int k0, int lane) {
  int r  = row0 + (lane & 15);
  int kh = (lane < 16) ? 0 : 8;
  const __bf16* p = base + (size_t)r * ld + (k0 + kh);
  V16U t;
  t.u[0] = *(const u32x4*)(p);        // K = kh+0..7
  t.u[1] = *(const u32x4*)(p + 16);   // K = kh+16..23
  return t.v;
}
// B-layout (32x16 bf16): lane holds column col0+(lane&15); lanes<16 hold K=k0+{0..7,16..23},
// lanes>=16 hold K=k0+8+{0..7,16..23}. Source must be "column-major addressable": base[col*ld + k].
__device__ __forceinline__ v16bf load_tileB(const __bf16* base, int ld, int col0, int k0, int lane) {
  int c  = col0 + (lane & 15);
  int kk = k0 + ((lane < 16) ? 0 : 8);
  const __bf16* p = base + (size_t)c * ld + kk;
  V16U t;
  t.u[0] = *(const u32x4*)(p);        // K = kk+0..7
  t.u[1] = *(const u32x4*)(p + 16);   // K = kk+16..23
  return t.v;
}

// ---------------- precision-conversion kernels ----------------
__global__ void cvt_f32_to_bf16(const float* __restrict__ src, __bf16* __restrict__ dst, int n) {
  for (int i = blockIdx.x * blockDim.x + threadIdx.x; i < n; i += gridDim.x * blockDim.x)
    dst[i] = f2bf(src[i]);
}
// wt[n*K + k] = w[k*N + n]  (transpose so GEMM B-operand is K-contiguous)
__global__ void transpose_to_bf16(const float* __restrict__ w, __bf16* __restrict__ wt, int K, int N) {
  int total = K * N;
  for (int i = blockIdx.x * blockDim.x + threadIdx.x; i < total; i += gridDim.x * blockDim.x) {
    int n = i / K, k = i - n * K;
    wt[i] = f2bf(w[(size_t)k * N + n]);
  }
}

// ---------------- QKV projection: [8192,1024] @ [1024,3072] ----------------
// Each wave: one 16(M) x 64(N) tile, K-loop of 32, software-pipelined (iter k+1
// operands loaded before iter k WMMAs so waits don't drain the load pipe).
// Writes Q (pre-scaled) [b,h,S,64], K [b,h,S,64], V transposed [b,h,64,S] in bf16.
__global__ void qkv_gemm(const __bf16* __restrict__ xb, const __bf16* __restrict__ wt,
                         __bf16* __restrict__ qb, __bf16* __restrict__ kb, __bf16* __restrict__ vt) {
  int wave = (blockIdx.x * blockDim.x + threadIdx.x) >> 5;
  int lane = threadIdx.x & 31;
  const int NT = (3 * H_) / 64;              // 48 N-tiles
  int mt = wave / NT, nt = wave - mt * NT;
  int m0 = mt * 16, n0 = nt * 64;

  v16bf a = load_tileA(xb, H_, m0, 0, lane);
  v16bf b[4];
#pragma unroll
  for (int t = 0; t < 4; ++t) b[t] = load_tileB(wt, H_, n0 + t * 16, 0, lane);

  v8f acc[4] = {};
  for (int k0 = 0; k0 < H_; k0 += 32) {
    int kn = (k0 + 32) & (H_ - 1);           // wrapped tail load (unused), keeps EXEC uniform
    v16bf an = load_tileA(xb, H_, m0, kn, lane);
    v16bf bn[4];
#pragma unroll
    for (int t = 0; t < 4; ++t) bn[t] = load_tileB(wt, H_, n0 + t * 16, kn, lane);
#pragma unroll
    for (int t = 0; t < 4; ++t) acc[t] = wmma_bf16(a, b[t], acc[t]);
    a = an;
#pragma unroll
    for (int t = 0; t < 4; ++t) b[t] = bn[t];
  }

#pragma unroll
  for (int t = 0; t < 4; ++t) {
    int col = n0 + t * 16 + (lane & 15);
#pragma unroll
    for (int i = 0; i < 8; ++i) {
      int row = m0 + i + ((lane < 16) ? 0 : 8);       // C layout: VGPR i -> M = i (+8 hi half)
      int bb = row >> 11, s = row & (S_ - 1);
      float val = acc[t][i];
      if (col < H_) {
        int h = col >> 6, d = col & 63;
        qb[(((size_t)(bb * NH_ + h)) * S_ + s) * HD_ + d] = f2bf(val * QK_SCALE);
      } else if (col < 2 * H_) {
        int c = col - H_; int h = c >> 6, d = c & 63;
        kb[(((size_t)(bb * NH_ + h)) * S_ + s) * HD_ + d] = f2bf(val);
      } else {
        int c = col - 2 * H_; int h = c >> 6, d = c & 63;
        vt[(((size_t)(bb * NH_ + h)) * HD_ + d) * S_ + s] = f2bf(val);
      }
    }
  }
}

// ---------------- fused masked flash attention ----------------
// One wave per (b,h,16-row q tile). Per 32-key chunk: issue ALL independent global
// loads (K tiles, V tiles, mask bytes) up front, then 4 score WMMAs, softmax VALU
// (overlaps outstanding V loads), LDS C->A relayout, 4 P*V WMMAs.
__global__ void attn_fused(const __bf16* __restrict__ qb, const __bf16* __restrict__ kb,
                           const __bf16* __restrict__ vt,
                           const unsigned char* __restrict__ amask,
                           const unsigned char* __restrict__ pmask,
                           __bf16* __restrict__ ctx) {
  __shared__ __align__(16) unsigned short ldsP[8][16][34];  // per-wave 16x32 P tile (padded)

  int widG = (blockIdx.x * blockDim.x + threadIdx.x) >> 5;
  int wloc = threadIdx.x >> 5;
  int lane = threadIdx.x & 31;

  int qt = widG & 127;            // S/16 = 128 q-tiles
  int bh = widG >> 7;             // 0..63
  int h = bh & 15, b = bh >> 4;
  int q0 = qt * 16;

  const __bf16* qp = qb + (size_t)bh * S_ * HD_;
  const __bf16* kp = kb + (size_t)bh * S_ * HD_;
  const __bf16* vp = vt + (size_t)bh * HD_ * S_;
  const unsigned char* am = amask + (size_t)b * S_ * S_;
  const unsigned char* pm = pmask + (size_t)b * S_;

  v16bf aq0 = load_tileA(qp, HD_, q0, 0, lane);   // Q rows, K-dim 0..31 (pre-scaled)
  v16bf aq1 = load_tileA(qp, HD_, q0, 32, lane);  // K-dim 32..63

  v8f o[4] = {};
  float mrow[8], lrow[8], padq[8];
  int myrow[8];
#pragma unroll
  for (int i = 0; i < 8; ++i) {
    mrow[i] = -1e30f; lrow[i] = 0.f;
    myrow[i] = q0 + i + ((lane < 16) ? 0 : 8);
    padq[i] = pm[myrow[i]] ? 1.f : 0.f;
  }

  for (int kc = 0; kc < S_; kc += 32) {
    // ---- prefetch next chunk of K and V into cache (global_prefetch_b8)
    int knx = (kc + 32) & (S_ - 1);
    __builtin_prefetch(kp + (size_t)(knx + lane) * HD_, 0, 3);        // 32 K rows
    __builtin_prefetch(vp + (size_t)(lane * 2)     * S_ + knx, 0, 3); // 64 V^T rows
    __builtin_prefetch(vp + (size_t)(lane * 2 + 1) * S_ + knx, 0, 3);

    // ---- issue ALL independent loads for this chunk up front
    v16bf b00 = load_tileB(kp, HD_, kc,      0,  lane);
    v16bf b01 = load_tileB(kp, HD_, kc,      32, lane);
    v16bf b10 = load_tileB(kp, HD_, kc + 16, 0,  lane);
    v16bf b11 = load_tileB(kp, HD_, kc + 16, 32, lane);
    v16bf bv[4];
#pragma unroll
    for (int d = 0; d < 4; ++d) bv[d] = load_tileB(vp, S_, d * 16, kc, lane);

    int k0i = kc + (lane & 15);
    int k1i = k0i + 16;
    float padk0 = pm[k0i] ? 1.f : 0.f;
    float padk1 = pm[k1i] ? 1.f : 0.f;
    float msk0[8], msk1[8];
#pragma unroll
    for (int i = 0; i < 8; ++i) {
      int r = myrow[i];
      float m0 = (am[(size_t)r * S_ + k0i] ? 1.f : 0.f) * padq[i] * padk0;
      float m1 = (am[(size_t)r * S_ + k1i] ? 1.f : 0.f) * padq[i] * padk1;
      if (r == k0i) m0 = 1.f;                  // | eye(S)
      if (r == k1i) m1 = 1.f;
      msk0[i] = m0; msk1[i] = m1;
    }

    // ---- scores: Q(16x64) @ K^T -> two 16x16 tiles
    v8f s0 = {}, s1 = {};
    s0 = wmma_bf16(aq0, b00, s0);
    s0 = wmma_bf16(aq1, b01, s0);
    s1 = wmma_bf16(aq0, b10, s1);
    s1 = wmma_bf16(aq1, b11, s1);

    // ---- online softmax (C layout: lane = key col within half, vgpr i = q row)
    float p0[8], p1[8];
#pragma unroll
    for (int i = 0; i < 8; ++i) {
      float v0 = (msk0[i] != 0.f) ? s0[i] : -1e30f;
      float v1 = (msk1[i] != 0.f) ? s1[i] : -1e30f;
      float cm = fmaxf(v0, v1);
      cm = fmaxf(cm, __shfl_xor(cm, 1, 16));   // row-max across the 16 lanes of this half
      cm = fmaxf(cm, __shfl_xor(cm, 2, 16));
      cm = fmaxf(cm, __shfl_xor(cm, 4, 16));
      cm = fmaxf(cm, __shfl_xor(cm, 8, 16));
      float nm = fmaxf(mrow[i], cm);
      float f  = __expf(mrow[i] - nm);
      p0[i] = msk0[i] * __expf(v0 - nm);
      p1[i] = msk1[i] * __expf(v1 - nm);
      float rs = p0[i] + p1[i];
      rs += __shfl_xor(rs, 1, 16);
      rs += __shfl_xor(rs, 2, 16);
      rs += __shfl_xor(rs, 4, 16);
      rs += __shfl_xor(rs, 8, 16);
      lrow[i] = lrow[i] * f + rs;
      mrow[i] = nm;
      o[0][i] *= f; o[1][i] *= f; o[2][i] *= f; o[3][i] *= f;
    }

    // ---- stage P (C layout) -> LDS -> reload in A layout (16x32 bf16)
#pragma unroll
    for (int i = 0; i < 8; ++i) {
      int r = i + ((lane < 16) ? 0 : 8);
      ldsP[wloc][r][lane & 15]        = f2bfbits(p0[i]);
      ldsP[wloc][r][16 + (lane & 15)] = f2bfbits(p1[i]);
    }
    asm volatile("s_wait_dscnt 0x0" ::: "memory");  // LDS in-order; fence compiler + hw
    V16U ap;
    {
      int r  = lane & 15;
      int kh = (lane < 16) ? 0 : 8;
#pragma unroll
      for (int j = 0; j < 8; ++j) {
        int kk = kh + 2 * (j & 3) + ((j >= 4) ? 16 : 0); // dword j -> A-layout K pair
        ap.w[j] = *(const unsigned int*)&ldsP[wloc][r][kk];
      }
    }
    // ---- O += P(16x32) @ V(32x64): 4 N-tiles of V^T (loads issued long ago)
#pragma unroll
    for (int d = 0; d < 4; ++d) o[d] = wmma_bf16(ap.v, bv[d], o[d]);
  }

  // ---- finalize: divide by row sums (diag guarantees lrow>0), store ctx bf16 [b,s,H]
#pragma unroll
  for (int t = 0; t < 4; ++t) {
    int d = t * 16 + (lane & 15);
#pragma unroll
    for (int i = 0; i < 8; ++i) {
      int r = myrow[i];
      float val = o[t][i] / lrow[i];
      ctx[((size_t)b * S_ + r) * H_ + h * HD_ + d] = f2bf(val);
    }
  }
}

// ---------------- output projection: ctx[8192,1024] @ w_out -> f32 out ----------------
__global__ void out_gemm(const __bf16* __restrict__ ctx, const __bf16* __restrict__ wot,
                         float* __restrict__ out) {
  int wave = (blockIdx.x * blockDim.x + threadIdx.x) >> 5;
  int lane = threadIdx.x & 31;
  const int NT = H_ / 64;                    // 16 N-tiles
  int mt = wave / NT, nt = wave - mt * NT;
  int m0 = mt * 16, n0 = nt * 64;

  v16bf a = load_tileA(ctx, H_, m0, 0, lane);
  v16bf b[4];
#pragma unroll
  for (int t = 0; t < 4; ++t) b[t] = load_tileB(wot, H_, n0 + t * 16, 0, lane);

  v8f acc[4] = {};
  for (int k0 = 0; k0 < H_; k0 += 32) {
    int kn = (k0 + 32) & (H_ - 1);
    v16bf an = load_tileA(ctx, H_, m0, kn, lane);
    v16bf bn[4];
#pragma unroll
    for (int t = 0; t < 4; ++t) bn[t] = load_tileB(wot, H_, n0 + t * 16, kn, lane);
#pragma unroll
    for (int t = 0; t < 4; ++t) acc[t] = wmma_bf16(a, b[t], acc[t]);
    a = an;
#pragma unroll
    for (int t = 0; t < 4; ++t) b[t] = bn[t];
  }

#pragma unroll
  for (int t = 0; t < 4; ++t) {
    int col = n0 + t * 16 + (lane & 15);
#pragma unroll
    for (int i = 0; i < 8; ++i) {
      int row = m0 + i + ((lane < 16) ? 0 : 8);
      out[(size_t)row * H_ + col] = acc[t][i];
    }
  }
}

// ---------------- launch ----------------
extern "C" void kernel_launch(void* const* d_in, const int* in_sizes, int n_in,
                              void* d_out, int out_size, void* d_ws, size_t ws_size,
                              hipStream_t stream) {
  const float*         x     = (const float*)d_in[0];
  const unsigned char* amask = (const unsigned char*)d_in[1]; // jax bool -> 1 byte
  const unsigned char* pmask = (const unsigned char*)d_in[2];
  const float*         w_qkv = (const float*)d_in[3];
  const float*         w_out = (const float*)d_in[4];
  float*               out   = (float*)d_out;

  // workspace carve-up (bf16 buffers, 256B aligned); total ~88 MB
  char* ws = (char*)d_ws;
  auto carve = [&](size_t bytes) -> void* {
    void* p = ws;
    ws += (bytes + 255) & ~(size_t)255;
    return p;
  };
  const size_t MS = (size_t)B_ * S_;           // 8192 rows
  __bf16* xb     = (__bf16*)carve(MS * H_ * 2);            // x in bf16
  __bf16* wqkv_t = (__bf16*)carve((size_t)3 * H_ * H_ * 2);// w_qkv^T [3072][1024]
  __bf16* wout_t = (__bf16*)carve((size_t)H_ * H_ * 2);    // w_out^T [1024][1024]
  __bf16* qb     = (__bf16*)carve(MS * H_ * 2);            // Q  [b,h,S,64] (pre-scaled)
  __bf16* kbuf   = (__bf16*)carve(MS * H_ * 2);            // K  [b,h,S,64]
  __bf16* vtb    = (__bf16*)carve(MS * H_ * 2);            // V^T[b,h,64,S]
  __bf16* ctx    = (__bf16*)carve(MS * H_ * 2);            // attention output bf16

  // 1) convert / transpose to bf16
  cvt_f32_to_bf16<<<4096, 256, 0, stream>>>(x, xb, (int)(MS * H_));
  transpose_to_bf16<<<2048, 256, 0, stream>>>(w_qkv, wqkv_t, H_, 3 * H_);
  transpose_to_bf16<<<1024, 256, 0, stream>>>(w_out, wout_t, H_, H_);

  // 2) QKV projection: 512 M-tiles x 48 N-tiles = 24576 waves, 8 waves/block
  qkv_gemm<<<3072, 256, 0, stream>>>(xb, wqkv_t, qb, kbuf, vtb);

  // 3) fused attention: 4*16*128 = 8192 waves
  attn_fused<<<1024, 256, 0, stream>>>(qb, kbuf, vtb, amask, pmask, ctx);

  // 4) output projection: 512 x 16 = 8192 waves
  out_gemm<<<1024, 256, 0, stream>>>(ctx, wout_t, out);
}
